// dispatcher_52175262712147
// MI455X (gfx1250) — compile-verified, matrix operand order checked by reading, not run on hardware
//
#include <hip/hip_runtime.h>
#include <math.h>

// ---------------------------------------------------------------------------
// Types for CDNA5 WMMA (wave32): V_WMMA_F32_16X16X32_BF16
// ---------------------------------------------------------------------------
typedef __attribute__((ext_vector_type(16))) __bf16        v16bf;
typedef __attribute__((ext_vector_type(8)))  float         v8f;
typedef __attribute__((ext_vector_type(4)))  unsigned int  u32x4;

__device__ __forceinline__ __bf16 f2bf(float f) {
  union { float f; unsigned int u; } in; in.f = f;
  unsigned int u = in.u;
  unsigned int r = (u + 0x7FFFu + ((u >> 16) & 1u)) >> 16;  // round-nearest-even
  union { unsigned short s; __bf16 h; } out; out.s = (unsigned short)r;
  return out.h;
}

// ---------------------------------------------------------------------------
// bf16 WMMA GEMM:  C[M,N] = A[M,K] @ W[N,K]^T + bias[N]
// Block: 256 threads (8 wave32), tile 128x128, K-step 32, double-buffered LDS.
// Each wave computes a 32x64 patch = 2x4 tiles of 16x16 (v8f accumulators).
// EPI 0: store bf16          (hidden activations h)
// EPI 1: store bf16 + inv_total += v, nxt -= v   (fused residual update)
// EPI 2: store f32           (loss-branch 'out')
// ---------------------------------------------------------------------------
#define BM 128
#define BN 128
#define BK 32
#define LDSS 40   // LDS row stride in bf16 elements (16B-aligned, conflict pad)

template <int EPI>
__global__ __launch_bounds__(256)
void gemm_bf16(const __bf16* __restrict__ A, const __bf16* __restrict__ W,
               const float* __restrict__ bias, int M, int N, int K,
               __bf16* __restrict__ obf, float* __restrict__ of32,
               float* __restrict__ inv_total, float* __restrict__ nxt)
{
  __shared__ __bf16 As[2][BM * LDSS];
  __shared__ __bf16 Ws[2][BM * LDSS];

  const int tid  = threadIdx.x;
  const int lane = tid & 31;
  const int wid  = tid >> 5;
  const int wm   = wid & 3;     // wave M offset: wm*32
  const int wn   = wid >> 2;    // wave N offset: wn*64
  const int hh   = lane >> 4;   // lane half (ISA 16-bit A/B layout)
  const int l15  = lane & 15;

  const int m0 = blockIdx.y * BM;
  const int n0 = blockIdx.x * BN;

  // Each thread stages 2 x 16B chunks for the A tile and 2 for the W tile.
  const int c0   = tid * 2;
  const int r0   = c0 >> 2,       kc0 = (c0 & 3) * 8;
  const int r1   = (c0 + 1) >> 2, kc1 = ((c0 + 1) & 3) * 8;

  const __bf16* Ag0 = A + (size_t)(m0 + r0) * K + kc0;
  const __bf16* Ag1 = A + (size_t)(m0 + r1) * K + kc1;
  const __bf16* Wg0 = W + (size_t)(n0 + r0) * K + kc0;
  const __bf16* Wg1 = W + (size_t)(n0 + r1) * K + kc1;

  const int sA0 = r0 * LDSS + kc0;
  const int sA1 = r1 * LDSS + kc1;

  v8f acc[2][4] = {};

  // Preload K-step 0
  *(u32x4*)&As[0][sA0] = *(const u32x4*)Ag0;
  *(u32x4*)&As[0][sA1] = *(const u32x4*)Ag1;
  *(u32x4*)&Ws[0][sA0] = *(const u32x4*)Wg0;
  *(u32x4*)&Ws[0][sA1] = *(const u32x4*)Wg1;
  __syncthreads();

  int cur = 0;
  for (int kk = 0; kk < K; kk += BK) {
    const bool more = (kk + BK) < K;
    u32x4 pa0, pa1, pw0, pw1;
    if (more) {                       // register prefetch of next K-step
      pa0 = *(const u32x4*)(Ag0 + kk + BK);
      pa1 = *(const u32x4*)(Ag1 + kk + BK);
      pw0 = *(const u32x4*)(Wg0 + kk + BK);
      pw1 = *(const u32x4*)(Wg1 + kk + BK);
    }

    // Fragment loads per ISA 16-bit layout:
    // lane half 0 holds K {0..7,16..23}, half 1 holds K {8..15,24..31}
    v16bf afrag[2], bfrag[4];
#pragma unroll
    for (int mt = 0; mt < 2; ++mt) {
      const __bf16* base = &As[cur][(wm * 32 + mt * 16 + l15) * LDSS + hh * 8];
      union { u32x4 u[2]; v16bf v; } t;
      t.u[0] = *(const u32x4*)(base);
      t.u[1] = *(const u32x4*)(base + 16);
      afrag[mt] = t.v;
    }
#pragma unroll
    for (int nt = 0; nt < 4; ++nt) {
      const __bf16* base = &Ws[cur][(wn * 64 + nt * 16 + l15) * LDSS + hh * 8];
      union { u32x4 u[2]; v16bf v; } t;
      t.u[0] = *(const u32x4*)(base);
      t.u[1] = *(const u32x4*)(base + 16);
      bfrag[nt] = t.v;
    }

#pragma unroll
    for (int mt = 0; mt < 2; ++mt)
#pragma unroll
      for (int nt = 0; nt < 4; ++nt)
        acc[mt][nt] = __builtin_amdgcn_wmma_f32_16x16x32_bf16(
            false, afrag[mt], false, bfrag[nt], (short)0, acc[mt][nt],
            false, false);

    if (more) {                       // stage prefetched data into other buffer
      *(u32x4*)&As[cur ^ 1][sA0] = pa0;
      *(u32x4*)&As[cur ^ 1][sA1] = pa1;
      *(u32x4*)&Ws[cur ^ 1][sA0] = pw0;
      *(u32x4*)&Ws[cur ^ 1][sA1] = pw1;
      __syncthreads();
      cur ^= 1;
    }
  }

  // Epilogue. C layout: VGPR e -> row (e + 8*half), col = lane&15.
#pragma unroll
  for (int mt = 0; mt < 2; ++mt) {
    const int gm0 = m0 + wm * 32 + mt * 16 + hh * 8;
#pragma unroll
    for (int nt = 0; nt < 4; ++nt) {
      const int gn = n0 + wn * 64 + nt * 16 + l15;
      const float bv = bias[gn];
#pragma unroll
      for (int e = 0; e < 8; ++e) {
        const float v = acc[mt][nt][e] + bv;
        const size_t off = (size_t)(gm0 + e) * N + gn;
        if (EPI == 0) {
          obf[off] = f2bf(v);
        } else if (EPI == 1) {
          obf[off] = f2bf(v);
          inv_total[off] += v;
          nxt[off]       -= v;
        } else {
          of32[off] = v;
        }
      }
    }
  }
}

// ---------------------------------------------------------------------------
// Helper kernels
// ---------------------------------------------------------------------------
__global__ __launch_bounds__(256)
void cvt_bf16(const float* __restrict__ in, __bf16* __restrict__ out, size_t n) {
  size_t i = (size_t)blockIdx.x * blockDim.x + threadIdx.x;
  const size_t stride = (size_t)gridDim.x * blockDim.x;
  for (; i < n; i += stride) out[i] = f2bf(in[i]);
}

__global__ __launch_bounds__(256)
void init_state(const float* __restrict__ x, float* __restrict__ inv_total,
                float* __restrict__ nxt, float* __restrict__ Dpart, size_t n) {
  size_t i = (size_t)blockIdx.x * blockDim.x + threadIdx.x;
  const size_t stride = (size_t)gridDim.x * blockDim.x;
  for (size_t j = i; j < n; j += stride) { inv_total[j] = 0.0f; nxt[j] = x[j]; }
  if (i < 8 * 1024) Dpart[i] = 0.0f;
}

// Per-row: S = sum(out), n2 = ||out||; c = out*S / max(|out|*sqrt(1024)*n2, eps).
// Done in place over the [8192,1024] buffer. One block per row.
__global__ __launch_bounds__(256)
void row_cos(float* __restrict__ buf) {
  __shared__ float rs[256], rq[256];
  const int t = threadIdx.x;
  float* p = buf + (size_t)blockIdx.x * 1024 + t * 4;
  float4 v = *(const float4*)p;
  rs[t] = v.x + v.y + v.z + v.w;
  rq[t] = v.x * v.x + v.y * v.y + v.z * v.z + v.w * v.w;
  __syncthreads();
  for (int st = 128; st > 0; st >>= 1) {
    if (t < st) { rs[t] += rs[t + st]; rq[t] += rq[t + st]; }
    __syncthreads();
  }
  const float S  = rs[0];
  const float n2 = sqrtf(rq[0]);
  float4 c;
  c.x = v.x * S / fmaxf(fabsf(v.x) * 32.0f * n2, 1e-8f);
  c.y = v.y * S / fmaxf(fabsf(v.y) * 32.0f * n2, 1e-8f);
  c.z = v.z * S / fmaxf(fabsf(v.z) * 32.0f * n2, 1e-8f);
  c.w = v.w * S / fmaxf(fabsf(v.w) * 32.0f * n2, 1e-8f);
  *(float4*)p = c;
}

// Column-wise sum over consecutive-row abs diffs, split into 8 deterministic
// row chunks (no atomics). Dpart[chunk][col] accumulated across iterations.
__global__ __launch_bounds__(256)
void coldiff(const float* __restrict__ c, float* __restrict__ Dpart) {
  const int col   = blockIdx.x * 256 + threadIdx.x;  // gridDim.x = 4
  const int chunk = blockIdx.y;                      // gridDim.y = 8
  const int r0 = chunk * 1024;
  int r1 = r0 + 1024; if (r1 > 8191) r1 = 8191;
  float prev = c[(size_t)r0 * 1024 + col];
  float acc  = 0.0f;
  for (int r = r0 + 1; r <= r1; ++r) {
    const float v = c[(size_t)r * 1024 + col];
    acc += fabsf(v - prev);
    prev = v;
  }
  Dpart[chunk * 1024 + col] += acc;
}

__global__ __launch_bounds__(256)
void loss_colsum(const float* __restrict__ Dpart, float* __restrict__ Dtot) {
  const int col = blockIdx.x * 256 + threadIdx.x;    // 4 blocks
  float s = 0.0f;
#pragma unroll
  for (int k = 0; k < 8; ++k) s += Dpart[k * 1024 + col];
  Dtot[col] = s;
}

__global__ __launch_bounds__(256)
void loss_broadcast(const float* __restrict__ Dtot, float* __restrict__ loss) {
  const size_t i = (size_t)blockIdx.x * 256 + threadIdx.x;  // 4096 blocks
  loss[i] = Dtot[i & 1023];
}

// ---------------------------------------------------------------------------
// Host launcher
// ---------------------------------------------------------------------------
extern "C" void kernel_launch(void* const* d_in, const int* in_sizes, int n_in,
                              void* d_out, int out_size, void* d_ws, size_t ws_size,
                              hipStream_t stream) {
  const float* x     = (const float*)d_in[0];
  const float* fc1W  = (const float*)d_in[1];
  const float* fc1b  = (const float*)d_in[2];
  const float* fc2W  = (const float*)d_in[3];
  const float* fc2b  = (const float*)d_in[4];
  const float* compW = (const float*)d_in[5];
  const float* compb = (const float*)d_in[6];

  const int Nr = 8192, D = 1024, H = 4096, L = 5;  // L = NUM_LAYERS + 1

  float* inv_total = (float*)d_out;
  float* nxt       = inv_total + (size_t)Nr * D;
  float* loss      = nxt + (size_t)Nr * D;

  char* w = (char*)d_ws;
  auto take = [&](size_t bytes) -> char* {
    char* p = w; w += (bytes + 255) & ~(size_t)255; return p;
  };
  __bf16* W1    = (__bf16*)take((size_t)H * D * 2);
  __bf16* W2    = (__bf16*)take((size_t)D * H * 2);
  __bf16* CW    = (__bf16*)take((size_t)L * D * D * 2);
  __bf16* Abf   = (__bf16*)take((size_t)Nr * D * 2);
  __bf16* Hbf   = (__bf16*)take((size_t)Nr * H * 2);
  __bf16* Ibf   = (__bf16*)take((size_t)Nr * D * 2);
  float*  Obuf  = (float*)take((size_t)Nr * D * 4);
  float*  Dpart = (float*)take(8 * 1024 * 4);
  float*  Dtot  = (float*)take(1024 * 4);

  // Weights -> bf16 once; initialize running state.
  cvt_bf16<<<2048, 256, 0, stream>>>(fc1W, W1, (size_t)H * D);
  cvt_bf16<<<2048, 256, 0, stream>>>(fc2W, W2, (size_t)D * H);
  cvt_bf16<<<2048, 256, 0, stream>>>(compW, CW, (size_t)L * D * D);
  init_state<<<4096, 256, 0, stream>>>(x, inv_total, nxt, Dpart, (size_t)Nr * D);

  for (int i = 0; i < L; ++i) {
    cvt_bf16<<<4096, 256, 0, stream>>>(nxt, Abf, (size_t)Nr * D);
    // h = nxt @ fc1_W^T + b1   [8192,4096]
    gemm_bf16<0><<<dim3(H / 128, Nr / 128), 256, 0, stream>>>(
        Abf, W1, fc1b, Nr, H, D, Hbf, nullptr, nullptr, nullptr);
    // inv = h @ fc2_W^T + b2;  inv_total += inv;  nxt -= inv   (fused)
    gemm_bf16<1><<<dim3(D / 128, Nr / 128), 256, 0, stream>>>(
        Hbf, W2, fc2b, Nr, D, H, Ibf, nullptr, inv_total, nxt);
    // out = inv @ comp_W[i]^T + comp_b[i]
    gemm_bf16<2><<<dim3(D / 128, Nr / 128), 256, 0, stream>>>(
        Ibf, CW + (size_t)i * D * D, compb + (size_t)i * D, Nr, D, D,
        nullptr, Obuf, nullptr, nullptr);
    // loss pieces (rank-1): c in place, then column abs-diff partials
    row_cos<<<Nr, 256, 0, stream>>>(Obuf);
    coldiff<<<dim3(4, 8), 256, 0, stream>>>(Obuf, Dpart);
  }

  loss_colsum<<<4, 256, 0, stream>>>(Dpart, Dtot);
  loss_broadcast<<<4096, 256, 0, stream>>>(Dtot, loss);
}
